// BasicLayer_25056839205741
// MI455X (gfx1250) — compile-verified, matrix operand order checked by reading, not run on hardware
//
#include <hip/hip_runtime.h>
#include <hip/hip_bf16.h>

// ---------------------------------------------------------------------------
// BasicLayer (windowed point-transformer, DEPTH=2) for MI455X / gfx1250.
// GEMMs: v_wmma_f32_16x16x32_bf16 (fp32 accum), weights pre-packed into the
// WMMA B-fragment layout and staged block-wide in LDS via async global->LDS
// (GLOBAL_LOAD_ASYNC_TO_LDS_B128, ASYNCcnt). Attention/softmax on VALU.
// ---------------------------------------------------------------------------

#define NPTS   49152
#define CCH    96
#define NH     6
#define HD     16
#define KWIN   12
#define NWIN   (NPTS / KWIN)          // 4096
#define PAIRS_PER_WIN (KWIN * KWIN)   // 144
#define TBL_LEN 320                   // 2*QGL

#if defined(__has_builtin)
#if __has_builtin(__builtin_amdgcn_global_load_async_to_lds_b128) && \
    __has_builtin(__builtin_amdgcn_s_wait_asynccnt)
#define HAVE_ASYNC_LDS 1
#endif
#endif
#ifndef HAVE_ASYNC_LDS
#define HAVE_ASYNC_LDS 0
#endif

typedef __attribute__((ext_vector_type(16))) __bf16 v16bf;
typedef __attribute__((ext_vector_type(8)))  float  v8f;

union ABu { v16bf v; uint4 q[2]; };

#if HAVE_ASYNC_LDS
// Toolchain's builtin takes typed 16-byte int-vector pointers
// (diagnostic: "__attribute__((__vector_size__(4 * sizeof(int)))) int *").
typedef int v4i_ __attribute__((vector_size(16)));
typedef __attribute__((address_space(1))) v4i_* as1_v4i;
typedef __attribute__((address_space(3))) v4i_* as3_v4i;
#endif

// K index inside a 32-wide chunk for lane-half L (0: lanes 0-15, 1: 16-31),
// element e (0..15), per the CDNA5 ISA 16-bit A/B fragment layout:
//   e 0..7  -> K = 8*L + e        (VGPR 0..3)
//   e 8..15 -> K = 16 + 8*L + e-8 (VGPR 4..7)
__device__ __forceinline__ int k_local(int e, int L) {
    return (e < 8) ? (8 * L + e) : (16 + 8 * L + (e - 8));
}

// ---------------------------------------------------------------------------
// Pack a K x Nc fp32 row-major weight matrix into bf16 WMMA B-fragments.
// Tile (kc, nt) occupies 512 contiguous bf16: lane*16 + e.
// ---------------------------------------------------------------------------
__global__ void pack_b_kernel(const float* __restrict__ W, __bf16* __restrict__ P,
                              int K, int Nc) {
    int idx = blockIdx.x * blockDim.x + threadIdx.x;
    int total = (K >> 5) * (Nc >> 4) * 512;
    if (idx >= total) return;
    int e    = idx & 15;
    int lane = (idx >> 4) & 31;
    int tile = idx >> 9;
    int ntiles = Nc >> 4;
    int kc = tile / ntiles;
    int nt = tile - kc * ntiles;
    int L = lane >> 4;
    int n = nt * 16 + (lane & 15);
    int k = kc * 32 + k_local(e, L);
    P[idx] = (__bf16)W[(size_t)k * Nc + n];
}

// ---------------------------------------------------------------------------
// d_out <- feats (fp32 copy; d_out is the residual-stream accumulator)
// ---------------------------------------------------------------------------
__global__ void copy_f32_kernel(const float* __restrict__ src, float* __restrict__ dst, int n) {
    int i = blockIdx.x * blockDim.x + threadIdx.x;
    if (i < n) dst[i] = src[i];
}

// ---------------------------------------------------------------------------
// LayerNorm over C=96, one wave per row, bf16 output (WMMA A-matrix source).
// ---------------------------------------------------------------------------
__global__ void ln_bf16_kernel(const float* __restrict__ X,
                               const float* __restrict__ g,
                               const float* __restrict__ b,
                               __bf16* __restrict__ out, int nrows) {
    int row = blockIdx.x * (blockDim.x >> 5) + (threadIdx.x >> 5);
    if (row >= nrows) return;
    int lane = threadIdx.x & 31;
    const float* xr = X + (size_t)row * CCH;
    float x0 = xr[lane], x1 = xr[32 + lane], x2 = xr[64 + lane];
    float s = x0 + x1 + x2;
    #pragma unroll
    for (int o = 16; o > 0; o >>= 1) s += __shfl_xor(s, o, 32);
    float mu = s * (1.0f / 96.0f);
    float d0 = x0 - mu, d1 = x1 - mu, d2 = x2 - mu;
    float vs = d0 * d0 + d1 * d1 + d2 * d2;
    #pragma unroll
    for (int o = 16; o > 0; o >>= 1) vs += __shfl_xor(vs, o, 32);
    float inv = rsqrtf(vs * (1.0f / 96.0f) + 1e-5f);
    __bf16* orow = out + (size_t)row * CCH;
    orow[lane]      = (__bf16)(d0 * inv * g[lane]      + b[lane]);
    orow[32 + lane] = (__bf16)(d1 * inv * g[32 + lane] + b[32 + lane]);
    orow[64 + lane] = (__bf16)(d2 * inv * g[64 + lane] + b[64 + lane]);
}

// ---------------------------------------------------------------------------
// WMMA GEMM: OUT(Mrows x Nc) = A(Mrows x K, bf16 row-major) @ Bp(packed) + bias
// Block = 8 waves = 8 consecutive M-tiles sharing ONE 32-wide N-pair whose
// packed B panel (all K chunks, <= 24 KB) is staged in LDS: async global->LDS
// when available, else load+ds_store. Inner loop: global_load_b128 (A) +
// ds_load_b128 (B) + 2x v_wmma_f32_16x16x32_bf16 per K chunk.
// MODE 0: store f32. MODE 1: accumulate (residual). MODE 2: GELU -> bf16.
// ---------------------------------------------------------------------------
template <int MODE>
__global__ void gemm_wmma_kernel(const __bf16* __restrict__ A,
                                 const __bf16* __restrict__ Bp,
                                 const float* __restrict__ bias,
                                 float* __restrict__ outf,
                                 __bf16* __restrict__ outbf,
                                 int Mrows, int K, int Nc) {
    __shared__ __align__(16) __bf16 Bsh[12 * 1024];   // up to 12 K-chunks * 2 tiles
    int npair = Nc >> 5;
    int mg = blockIdx.x / npair;
    int np = blockIdx.x - mg * npair;
    int wv   = threadIdx.x >> 5;
    int lane = threadIdx.x & 31;
    int kchunks = K >> 5;
    int ntiles  = Nc >> 4;

    // --- stage the B panel for this N-pair (all K chunks) into LDS --------
    int units = kchunks << 7;                         // 16-byte units
    for (int u = threadIdx.x; u < units; u += blockDim.x) {
        int kc = u >> 7;
        int wi = u & 127;
        size_t src = ((size_t)(kc * ntiles + np * 2)) * 512 + (size_t)wi * 8;
        int    dst = kc * 1024 + wi * 8;
#if HAVE_ASYNC_LDS
        __builtin_amdgcn_global_load_async_to_lds_b128(
            (as1_v4i)(uintptr_t)(const void*)(Bp + src),
            (as3_v4i)(uintptr_t)(void*)(&Bsh[dst]), 0, 0);
#else
        *(uint4*)&Bsh[dst] = *(const uint4*)&Bp[src];
#endif
    }
#if HAVE_ASYNC_LDS
    __builtin_amdgcn_s_wait_asynccnt(0);
#endif
    __syncthreads();

    int mt = mg * 8 + wv;
    if (mt * 16 >= Mrows) return;
    int l16 = lane & 15, L = lane >> 4;
    int m = mt * 16 + l16;

    v8f acc0 = {};
    v8f acc1 = {};
    const __bf16* arow = A + (size_t)m * K;
    for (int kc = 0; kc < kchunks; ++kc) {
        ABu a, b0, b1;
        const __bf16* abase = arow + kc * 32 + 8 * L;
        a.q[0] = *(const uint4*)(abase);
        a.q[1] = *(const uint4*)(abase + 16);         // +16 elems = +32 bytes
        const __bf16* bb = &Bsh[kc * 1024 + lane * 16];
        b0.q[0] = *(const uint4*)(bb);
        b0.q[1] = *(const uint4*)(bb + 8);
        b1.q[0] = *(const uint4*)(bb + 512);
        b1.q[1] = *(const uint4*)(bb + 520);
        if (kc + 1 < kchunks) __builtin_prefetch(abase + 32, 0, 1);
        acc0 = __builtin_amdgcn_wmma_f32_16x16x32_bf16(false, a.v, false, b0.v,
                                                       (short)0, acc0, false, false);
        acc1 = __builtin_amdgcn_wmma_f32_16x16x32_bf16(false, a.v, false, b1.v,
                                                       (short)0, acc1, false, false);
    }

    int nbase = np * 32;
    float bi0 = bias[nbase + l16];
    float bi1 = bias[nbase + 16 + l16];
    #pragma unroll
    for (int r = 0; r < 8; ++r) {
        int mm = mt * 16 + r + 8 * L;    // C/D layout: VGPR r holds M = r + 8*L
        float v0 = acc0[r] + bi0;
        float v1 = acc1[r] + bi1;
        size_t o0 = (size_t)mm * Nc + nbase + l16;
        size_t o1 = o0 + 16;
        if (MODE == 0) {
            outf[o0] = v0; outf[o1] = v1;
        } else if (MODE == 1) {
            outf[o0] += v0; outf[o1] += v1;
        } else {
            float g0 = 0.5f * v0 * (1.0f + erff(v0 * 0.70710678118654752f));
            float g1 = 0.5f * v1 * (1.0f + erff(v1 * 0.70710678118654752f));
            outbf[o0] = (__bf16)g0; outbf[o1] = (__bf16)g1;
        }
    }
}

// ---------------------------------------------------------------------------
// Windowed attention: one block per window (12 points), one thread per
// (row, head). qk dots + quantized-rel-pos table biases, 12-wide softmax,
// weighted v sum. Output bf16 (A-matrix for the proj GEMM).
// qkv layout: [p][s][h][d] at p*288 + s*96 + h*16 + d.
// tq/tk layout: [ridx][h][d][dim] at ((ridx*6 + h)*16 + d)*3 + dim.
// ---------------------------------------------------------------------------
__global__ void attn_kernel(const float* __restrict__ qkv,
                            const float* __restrict__ xyz,
                            const int*   __restrict__ i1,
                            const float* __restrict__ tq,
                            const float* __restrict__ tk,
                            __bf16* __restrict__ outbf) {
    __shared__ int   mem[KWIN];
    __shared__ float sxyz[KWIN][3];
    int w = blockIdx.x;
    int t = threadIdx.x;
    if (t < KWIN) {
        int p = i1[(size_t)w * PAIRS_PER_WIN + t];
        mem[t] = p;
        sxyz[t][0] = xyz[(size_t)p * 3 + 0];
        sxyz[t][1] = xyz[(size_t)p * 3 + 1];
        sxyz[t][2] = xyz[(size_t)p * 3 + 2];
    }
    __syncthreads();
    if (t >= KWIN * NH) return;
    int r = t / NH, h = t - r * NH;
    int pr = mem[r];

    float qv[HD];
    const float* qp = qkv + (size_t)pr * 288 + h * HD;          // s=0
    #pragma unroll
    for (int d = 0; d < HD; ++d) qv[d] = qp[d] * 0.25f;          // SCALE = 16^-0.5

    float attn[KWIN];
    for (int c = 0; c < KWIN; ++c) {
        int pc = mem[c];
        const float* kp = qkv + (size_t)pc * 288 + 96 + h * HD;  // s=1
        float kv[HD];
        float dot = 0.0f;
        #pragma unroll
        for (int d = 0; d < HD; ++d) { kv[d] = kp[d]; dot += qv[d] * kv[d]; }
        float bias = 0.0f;
        #pragma unroll
        for (int dim = 0; dim < 3; ++dim) {
            float rp = sxyz[r][dim] - sxyz[c][dim];
            rp = rintf(rp * 100000.0f) * (1.0f / 100000.0f);     // round half-even
            int ridx = (int)floorf((rp + 1.6f - 0.0001f) * 100.0f);
            const float* tqp = tq + (((size_t)ridx * NH + h) * HD) * 3 + dim;
            const float* tkp = tk + (((size_t)ridx * NH + h) * HD) * 3 + dim;
            #pragma unroll
            for (int d = 0; d < HD; ++d)
                bias += qv[d] * tqp[d * 3] + kv[d] * tkp[d * 3];
        }
        attn[c] = dot + bias;
    }

    float mx = attn[0];
    #pragma unroll
    for (int c = 1; c < KWIN; ++c) mx = fmaxf(mx, attn[c]);
    float den = 0.0f;
    #pragma unroll
    for (int c = 0; c < KWIN; ++c) { attn[c] = __expf(attn[c] - mx); den += attn[c]; }
    float inv = 1.0f / den;

    float o[HD];
    #pragma unroll
    for (int d = 0; d < HD; ++d) o[d] = 0.0f;
    for (int c = 0; c < KWIN; ++c) {
        const float* vp = qkv + (size_t)mem[c] * 288 + 192 + h * HD;  // s=2
        float sm = attn[c] * inv;
        #pragma unroll
        for (int d = 0; d < HD; ++d) o[d] += sm * vp[d];
    }
    __bf16* op = outbf + (size_t)pr * CCH + h * HD;
    #pragma unroll
    for (int d = 0; d < HD; ++d) op[d] = (__bf16)o[d];
}

// ---------------------------------------------------------------------------
// host launcher
// ---------------------------------------------------------------------------
extern "C" void kernel_launch(void* const* d_in, const int* in_sizes, int n_in,
                              void* d_out, int out_size, void* d_ws, size_t ws_size,
                              hipStream_t stream) {
    const float* feats   = (const float*)d_in[0];
    const float* xyz     = (const float*)d_in[1];
    const int*   i1_reg  = (const int*)  d_in[3];
    const int*   i1_shf  = (const int*)  d_in[5];
    const float* ln1_g   = (const float*)d_in[6];
    const float* ln1_b   = (const float*)d_in[7];
    const float* qkv_w   = (const float*)d_in[8];
    const float* qkv_b   = (const float*)d_in[9];
    const float* proj_w  = (const float*)d_in[10];
    const float* proj_b  = (const float*)d_in[11];
    const float* ln2_g   = (const float*)d_in[12];
    const float* ln2_b   = (const float*)d_in[13];
    const float* fc1_w   = (const float*)d_in[14];
    const float* fc1_b   = (const float*)d_in[15];
    const float* fc2_w   = (const float*)d_in[16];
    const float* fc2_b   = (const float*)d_in[17];
    const float* table_q = (const float*)d_in[18];
    const float* table_k = (const float*)d_in[19];

    char* ws = (char*)d_ws;
    size_t off = 0;
    float*  qkv_f32   = (float*)(ws + off);  off += (size_t)NPTS * 288 * sizeof(float);   // 56.6 MB
    __bf16* xln_bf16  = (__bf16*)(ws + off); off += (size_t)NPTS * CCH * 2;               //  9.4 MB (ln1 & ln2)
    __bf16* attn_bf16 = (__bf16*)(ws + off); off += (size_t)NPTS * CCH * 2;               //  9.4 MB
    __bf16* gelu_bf16 = (__bf16*)(ws + off); off += (size_t)NPTS * 384 * 2;               // 37.7 MB
    __bf16* qkvP      = (__bf16*)(ws + off); off += (size_t)2 * 96 * 288 * 2;
    __bf16* projP     = (__bf16*)(ws + off); off += (size_t)2 * 96 * 96  * 2;
    __bf16* fc1P      = (__bf16*)(ws + off); off += (size_t)2 * 96 * 384 * 2;
    __bf16* fc2P      = (__bf16*)(ws + off); off += (size_t)2 * 384 * 96 * 2;

    float* acc = (float*)d_out;  // residual stream

    // --- pack weights (both depths) ------------------------------------
    for (int d = 0; d < 2; ++d) {
        pack_b_kernel<<<(96 * 288 + 255) / 256, 256, 0, stream>>>(
            qkv_w + (size_t)d * 96 * 288, qkvP + (size_t)d * 96 * 288, 96, 288);
        pack_b_kernel<<<(96 * 96 + 255) / 256, 256, 0, stream>>>(
            proj_w + (size_t)d * 96 * 96, projP + (size_t)d * 96 * 96, 96, 96);
        pack_b_kernel<<<(96 * 384 + 255) / 256, 256, 0, stream>>>(
            fc1_w + (size_t)d * 96 * 384, fc1P + (size_t)d * 96 * 384, 96, 384);
        pack_b_kernel<<<(384 * 96 + 255) / 256, 256, 0, stream>>>(
            fc2_w + (size_t)d * 384 * 96, fc2P + (size_t)d * 384 * 96, 384, 96);
    }

    // --- residual stream init -----------------------------------------
    copy_f32_kernel<<<(NPTS * CCH + 255) / 256, 256, 0, stream>>>(feats, acc, NPTS * CCH);

    const int lnBlocks = NPTS / 8;         // 8 rows per 256-thread block
    const int mgroups  = (NPTS / 16) / 8;  // 384 blocks of 8 M-tiles

    for (int d = 0; d < 2; ++d) {
        const int* i1 = (d == 0) ? i1_reg : i1_shf;

        // LN1 -> bf16
        ln_bf16_kernel<<<lnBlocks, 256, 0, stream>>>(
            acc, ln1_g + d * CCH, ln1_b + d * CCH, xln_bf16, NPTS);

        // qkv GEMM: N x 96 @ 96 x 288 -> f32
        gemm_wmma_kernel<0><<<mgroups * (288 / 32), 256, 0, stream>>>(
            xln_bf16, qkvP + (size_t)d * 96 * 288, qkv_b + d * 288,
            qkv_f32, (__bf16*)nullptr, NPTS, 96, 288);

        // windowed attention -> bf16
        attn_kernel<<<NWIN, 96, 0, stream>>>(
            qkv_f32, xyz, i1,
            table_q + (size_t)d * TBL_LEN * NH * HD * 3,
            table_k + (size_t)d * TBL_LEN * NH * HD * 3,
            attn_bf16);

        // proj GEMM + residual: acc += attn @ (96x96) + b
        gemm_wmma_kernel<1><<<mgroups * (96 / 32), 256, 0, stream>>>(
            attn_bf16, projP + (size_t)d * 96 * 96, proj_b + d * CCH,
            acc, (__bf16*)nullptr, NPTS, 96, 96);

        // LN2 -> bf16 (reuse xln buffer)
        ln_bf16_kernel<<<lnBlocks, 256, 0, stream>>>(
            acc, ln2_g + d * CCH, ln2_b + d * CCH, xln_bf16, NPTS);

        // fc1 GEMM + GELU -> bf16
        gemm_wmma_kernel<2><<<mgroups * (384 / 32), 256, 0, stream>>>(
            xln_bf16, fc1P + (size_t)d * 96 * 384, fc1_b + d * 384,
            (float*)nullptr, gelu_bf16, NPTS, 96, 384);

        // fc2 GEMM + residual: acc += gelu @ (384x96) + b
        gemm_wmma_kernel<1><<<mgroups * (96 / 32), 256, 0, stream>>>(
            gelu_bf16, fc2P + (size_t)d * 384 * 96, fc2_b + d * CCH,
            acc, (__bf16*)nullptr, NPTS, 384, 96);
    }
    (void)in_sizes; (void)n_in; (void)out_size; (void)ws_size;
}
